// GeometricBlock_40046275067961
// MI455X (gfx1250) — matrix operand order repaired and checked by reading, not compiled, not run on hardware
//
#include <hip/hip_runtime.h>
#include <hip/hip_bf16.h>

typedef __attribute__((ext_vector_type(16))) __bf16 v16bf;
typedef __attribute__((ext_vector_type(8)))  __bf16 v8bf;
typedef __attribute__((ext_vector_type(8)))  float  v8f;
typedef __attribute__((ext_vector_type(4)))  float  v4f;

#define N_NODES 20000
#define N_PAIRS 640000
#define FDIM    128
#define KRBF    32
#define M_TOT   16

// packed bf16 weight image in d_ws (elements):
//   [0,4096)      W1r  [n][k]  128x32
//   [4096,8192)   W1s  [n][k]  128x32 (K padded 4->32 with zeros)
//   [8192,24576)  W2r  [n][k]  128x128
//   [24576,40960) W2s  [n][k]  128x128
#define WBF_ELEMS 40960
#define WBF_BYTES (WBF_ELEMS * 2)      // 81920

__device__ __forceinline__ v8f wmma_bf16(v16bf a, v16bf b, v8f c) {
    return __builtin_amdgcn_wmma_f32_16x16x32_bf16(
        false, a, false, b, (short)0, c, false, false);
}

__device__ __forceinline__ v16bf cat2(v8bf lo, v8bf hi) {
    v16bf r;
#pragma unroll
    for (int i = 0; i < 8; ++i) { r[i] = lo[i]; r[8 + i] = hi[i]; }
    return r;
}

__device__ __forceinline__ v8bf cvt8g(const float* p) {
    v4f a = *(const v4f*)p;
    v4f b = *(const v4f*)(p + 4);
    v8bf r;
#pragma unroll
    for (int i = 0; i < 4; ++i) { r[i] = (__bf16)a[i]; r[4 + i] = (__bf16)b[i]; }
    return r;
}

__device__ __forceinline__ float silu(float z) { return z / (1.f + __expf(-z)); }

// -------------------------------------------------------------------------
__global__ void zero_kernel(float* __restrict__ out, int n) {
    int i = blockIdx.x * blockDim.x + threadIdx.x;
    if (i < n) out[i] = 0.f;
}

// -------------------------------------------------------------------------
// one-shot: transpose + pad + cast weights to a bf16 image the pair kernel
// can DMA straight into LDS with async copies (no per-WG convert work).
__global__ void wprep_kernel(const float* __restrict__ radW1,
                             const float* __restrict__ sphW1,
                             const float* __restrict__ radW2,
                             const float* __restrict__ sphW2,
                             __bf16* __restrict__ wbf) {
    int i = blockIdx.x * blockDim.x + threadIdx.x;
    if (i < 4096) {
        int n = i >> 5, k = i & 31;
        wbf[i] = (__bf16)radW1[k * 128 + n];
    } else if (i < 8192) {
        int j = i - 4096, n = j >> 5, k = j & 31;
        wbf[i] = (__bf16)((k < 4) ? sphW1[k * 128 + n] : 0.f);
    } else if (i < 24576) {
        int j = i - 8192, n = j >> 7, k = j & 127;
        wbf[i] = (__bf16)radW2[k * 128 + n];
    } else if (i < WBF_ELEMS) {
        int j = i - 24576, n = j >> 7, k = j & 127;
        wbf[i] = (__bf16)sphW2[k * 128 + n];
    }
}

// -------------------------------------------------------------------------
__global__ __launch_bounds__(256, 1)
void qk_kernel(const float* __restrict__ x,
               const float* __restrict__ Wq, const float* __restrict__ Wk,
               float* __restrict__ qws, float* __restrict__ kws) {
    __shared__ __bf16 sQ[4 * 32 * 32];   // [h][g(n)][f(k)]
    __shared__ __bf16 sK[4 * 32 * 32];
    const int tid  = threadIdx.x;
    const int wid  = tid >> 5;
    const int lane = tid & 31;

    for (int i = tid; i < 4096; i += 256) {
        int h = i >> 10, rem = i & 1023, g = rem >> 5, f = rem & 31;
        sQ[i] = (__bf16)Wq[h * 1024 + f * 32 + g];
        sK[i] = (__bf16)Wk[h * 1024 + f * 32 + g];
    }
    __syncthreads();

    const int row0 = blockIdx.x * 128 + wid * 16;
    const int mrow = lane & 15;
    const int half = lane >> 4;
    int rowA = row0 + mrow;
    if (rowA >= N_NODES) rowA = N_NODES - 1;   // clamp; EXEC stays all-1s for WMMA
    const int kA = half ? 16 : 0;
    const int kB = half ? 16 : 0;
    const bool fullTile = (row0 + 16 <= N_NODES);

#pragma unroll
    for (int h = 0; h < 4; ++h) {
        v16bf a = cat2(cvt8g(x + (size_t)rowA * FDIM + h * 32 + kA),
                       cvt8g(x + (size_t)rowA * FDIM + h * 32 + kA + 8));
#pragma unroll
        for (int t = 0; t < 2; ++t) {
            const int n = t * 16 + mrow;
            v16bf bq = *(const v16bf*)(&sQ[h * 1024 + n * 32 + kB]);
            v16bf bk = *(const v16bf*)(&sK[h * 1024 + n * 32 + kB]);
            v8f cq = {}; cq = wmma_bf16(a, bq, cq);
            v8f ck = {}; ck = wmma_bf16(a, bk, ck);
#pragma unroll
            for (int r = 0; r < 8; ++r) {
                const int m = r + half * 8;
                const int grow = row0 + m;
                if (fullTile || grow < N_NODES) {
                    qws[(size_t)grow * FDIM + h * 32 + n] = cq[r];
                    kws[(size_t)grow * FDIM + h * 32 + n] = ck[r];
                }
            }
        }
    }
}

// -------------------------------------------------------------------------
// One workgroup = 128 pairs; 8 waves x 16 rows.  Weights DMA'd into LDS with
// CDNA5 async global->LDS copies; MLPs via bf16 WMMA; alpha + atomic scatter.
__global__ __launch_bounds__(256, 1)
void pair_kernel(const float* __restrict__ sph,
                 const float* __restrict__ rbf,
                 const float* __restrict__ dchi,
                 const float* __restrict__ phi_r,
                 const float* __restrict__ phi_chi,
                 const float* __restrict__ mask,
                 const int*   __restrict__ idx_i,
                 const int*   __restrict__ idx_j,
                 const float* __restrict__ radb1, const float* __restrict__ radb2,
                 const float* __restrict__ sphb1, const float* __restrict__ sphb2,
                 const __bf16* __restrict__ wbf,
                 const float* __restrict__ qws,   const float* __restrict__ kws,
                 float* __restrict__ out) {
    // manual LDS carve-up: one base => simple async-copy addressing
    __shared__ __attribute__((aligned(16))) unsigned char smem[WBF_BYTES + 8 * 2048 * 4];
    __bf16* sWall = (__bf16*)smem;                    // 40960 bf16 weight image
    __bf16* sW1r  = sWall;                            // [n*32 + k]
    __bf16* sW1s  = sWall + 4096;
    __bf16* sW2r  = sWall + 8192;                     // [n*128 + k]
    __bf16* sW2s  = sWall + 24576;
    float*  swb   = (float*)(smem + WBF_BYTES);       // 8 waves x 16x128 f32

    const int tid  = threadIdx.x;
    const int wid  = tid >> 5;
    const int lane = tid & 31;

    // ---- async DMA of the bf16 weight image: global -> LDS, no VGPR staging
    {
        const unsigned int lds_base = (unsigned int)(uintptr_t)(void*)smem;  // addr[31:0] = LDS offset
        const unsigned long long gbase = (unsigned long long)(uintptr_t)wbf;
        for (int i = tid; i < WBF_BYTES / 16; i += 256) {
            unsigned int       ldsa = lds_base + (unsigned int)(i * 16);
            unsigned long long ga   = gbase + (unsigned long long)(i * 16);
            asm volatile("global_load_async_to_lds_b128 %0, %1, off"
                         :: "v"(ldsa), "v"(ga) : "memory");
        }
        asm volatile("s_wait_asynccnt 0x0" ::: "memory");
    }

    const int pair0 = blockIdx.x * 128 + wid * 16;
    const int mrow  = lane & 15;
    const int half  = lane >> 4;
    const int row   = pair0 + mrow;      // P divisible by 128: no edge cases
    const int kA    = half ? 16 : 0;
    const int kB    = half ? 16 : 0;
    const int ncol  = mrow;

    // prefetch next block's pair-stream rows while WMMAs run
    if (blockIdx.x + 1 < N_PAIRS / 128) {
        const int nrow = row + 128;
        __builtin_prefetch(rbf + (size_t)nrow * KRBF, 0, 1);
        __builtin_prefetch(sph + (size_t)nrow * M_TOT, 0, 1);
    }

    // A fragment: rbf row (16x32)
    v16bf aR = cat2(cvt8g(rbf + (size_t)row * KRBF + kA),
                    cvt8g(rbf + (size_t)row * KRBF + kA + 8));
    // A fragment: d_chi row zero-padded (16x32)
    v16bf aS;
    {
        v8bf z;
#pragma unroll
        for (int i = 0; i < 8; ++i) z[i] = (__bf16)0.f;
        if (half == 0) {
            v4f d = *(const v4f*)(dchi + (size_t)row * 4);
            v8bf lo;
#pragma unroll
            for (int i = 0; i < 4; ++i) { lo[i] = (__bf16)d[i]; lo[4 + i] = (__bf16)0.f; }
            aS = cat2(lo, z);
        } else {
            aS = cat2(z, z);
        }
    }

    __syncthreads();   // all waves' async weight copies complete + visible

    __bf16* hbuf = (__bf16*)(&swb[wid * 2048]);   // 16x128 bf16 view
    float*  wbuf = &swb[wid * 2048];              // 16x128 f32 view
    v8f wacc[8];

    // ---------------- RAD path ----------------
#pragma unroll
    for (int t = 0; t < 8; ++t) {                    // layer 1: K=32, 1 WMMA
        v8f c = {};
        v16bf b = *(const v16bf*)(&sW1r[(t * 16 + ncol) * 32 + kB]);
        c = wmma_bf16(aR, b, c);
        const int n = t * 16 + ncol;
        const float b1 = radb1[n];
#pragma unroll
        for (int r = 0; r < 8; ++r)
            hbuf[(r + half * 8) * 128 + n] = (__bf16)silu(c[r] + b1);
    }
#pragma unroll
    for (int t = 0; t < 8; ++t) {                    // layer 2: K=128, 4 WMMA
        v8f c = {};
#pragma unroll
        for (int s = 0; s < 4; ++s) {
            v16bf a = cat2(*(const v8bf*)(&hbuf[mrow * 128 + s * 32 + kA]),
                           *(const v8bf*)(&hbuf[mrow * 128 + s * 32 + kA + 8]));
            v16bf b = *(const v16bf*)(&sW2r[(t * 16 + ncol) * 128 + s * 32 + kB]);
            c = wmma_bf16(a, b, c);
        }
        wacc[t] = c;
    }

    // ---------------- SPH path (reuses hbuf) ----------------
#pragma unroll
    for (int t = 0; t < 8; ++t) {
        v8f c = {};
        v16bf b = *(const v16bf*)(&sW1s[(t * 16 + ncol) * 32 + kB]);
        c = wmma_bf16(aS, b, c);
        const int n = t * 16 + ncol;
        const float b1 = sphb1[n];
#pragma unroll
        for (int r = 0; r < 8; ++r)
            hbuf[(r + half * 8) * 128 + n] = (__bf16)silu(c[r] + b1);
    }
#pragma unroll
    for (int t = 0; t < 8; ++t) {
        v8f c = {};
#pragma unroll
        for (int s = 0; s < 4; ++s) {
            v16bf a = cat2(*(const v8bf*)(&hbuf[mrow * 128 + s * 32 + kA]),
                           *(const v8bf*)(&hbuf[mrow * 128 + s * 32 + kA + 8]));
            v16bf b = *(const v16bf*)(&sW2s[(t * 16 + ncol) * 128 + s * 32 + kB]);
            c = wmma_bf16(a, b, c);
        }
        wacc[t] = wacc[t] + c;
    }

    // w_ij (+ biases; pair_mask folded into alpha) -> f32 LDS
#pragma unroll
    for (int t = 0; t < 8; ++t) {
        const int n = t * 16 + ncol;
        const float bb = radb2[n] + sphb2[n];
#pragma unroll
        for (int r = 0; r < 8; ++r)
            wbuf[(r + half * 8) * 128 + n] = wacc[t][r] + bb;
    }

    // ---------------- alpha ----------------
    float alph[2];
#pragma unroll
    for (int cc = 0; cc < 2; ++cc) {
        const int c = lane + 32 * cc;    // c = p*4 + h
        const int p = c >> 2, h = c & 3;
        const int pr = pair0 + p;
        const int gi = idx_i[pr], gj = idx_j[pr];
        const float* qr = qws + (size_t)gi * FDIM + h * 32;
        const float* kr = kws + (size_t)gj * FDIM + h * 32;
        const float* wr = wbuf + p * 128 + h * 32;
        float acc = 0.f;
#pragma unroll
        for (int f = 0; f < 32; ++f) acc += qr[f] * wr[f] * kr[f];
        const float mk = mask[pr];
        alph[cc] = acc * 0.17677669529663689f * mk * mk * (phi_r[pr] + phi_chi[pr]);
    }
    wbuf[lane]      = alph[0];   // same-wave DS ops stay in order
    wbuf[lane + 32] = alph[1];

    // ---------------- scatter ----------------
#pragma unroll
    for (int j = 0; j < 8; ++j) {
        const int task = lane + 32 * j;          // 16 pairs x 16 m
        const int p = task >> 4, m = task & 15;
        const int pr = pair0 + p;
        const int hm = (m == 0) ? 0 : (m < 4 ? 1 : (m < 9 ? 2 : 3));   // repeats 1,3,5,7
        const float a = wbuf[p * 4 + hm];
        const float v = a * sph[(size_t)pr * M_TOT + m];
        atomicAdd(out + (size_t)idx_i[pr] * M_TOT + m, v);
    }
}

// -------------------------------------------------------------------------
extern "C" void kernel_launch(void* const* d_in, const int* in_sizes, int n_in,
                              void* d_out, int out_size, void* d_ws, size_t ws_size,
                              hipStream_t stream) {
    const float* sph     = (const float*)d_in[1];
    const float* x       = (const float*)d_in[2];
    const float* rbf     = (const float*)d_in[3];
    const float* dchi    = (const float*)d_in[4];
    const float* phi_r   = (const float*)d_in[5];
    const float* phi_chi = (const float*)d_in[6];
    const float* mask    = (const float*)d_in[7];
    const int*   idx_i   = (const int*)d_in[8];
    const int*   idx_j   = (const int*)d_in[9];
    const float* radW1   = (const float*)d_in[10];
    const float* radb1   = (const float*)d_in[11];
    const float* radW2   = (const float*)d_in[12];
    const float* radb2   = (const float*)d_in[13];
    const float* sphW1   = (const float*)d_in[14];
    const float* sphb1   = (const float*)d_in[15];
    const float* sphW2   = (const float*)d_in[16];
    const float* sphb2   = (const float*)d_in[17];
    const float* Wq      = (const float*)d_in[18];
    const float* Wk      = (const float*)d_in[19];

    float*  out = (float*)d_out;
    float*  qws = (float*)d_ws;                         // 20000*128 f32
    float*  kws = qws + (size_t)N_NODES * FDIM;         // 20000*128 f32
    __bf16* wbf = (__bf16*)(kws + (size_t)N_NODES * FDIM);  // 40960 bf16

    zero_kernel<<<(N_NODES * M_TOT + 255) / 256, 256, 0, stream>>>(out, N_NODES * M_TOT);
    wprep_kernel<<<(WBF_ELEMS + 255) / 256, 256, 0, stream>>>(radW1, sphW1, radW2, sphW2, wbf);
    qk_kernel<<<(N_NODES + 127) / 128, 256, 0, stream>>>(x, Wq, Wk, qws, kws);
    pair_kernel<<<N_PAIRS / 128, 256, 0, stream>>>(
        sph, rbf, dchi, phi_r, phi_chi, mask, idx_i, idx_j,
        radb1, radb2, sphb1, sphb2, wbf, qws, kws, out);
}
